// SplineNet_24464133718759
// MI455X (gfx1250) — matrix operand order repaired and checked by reading, not compiled, not run on hardware
//
#include <hip/hip_runtime.h>

// ---------------------------------------------------------------------------
// SplineNet (2x SplineConv + MLP head) for MI455X / gfx1250.
// Scatter: f32 atomics (memory-bound phase). Dense contractions: bf16 WMMA
// (v_wmma_f32_16x16x32_bf16) with f32 accumulation.
// ---------------------------------------------------------------------------

typedef __attribute__((ext_vector_type(16))) __bf16 v16bf;
typedef __attribute__((ext_vector_type(8)))  float  v8f;

#define N_NODES 50000
#define N_EDGES 1600000
#define DIM     64

// float -> bf16 via native conversion (lowers to v_cvt_pk_bf16_f32 on gfx1250;
// consecutive pairs get fused into one packed convert)
__device__ __forceinline__ __bf16 f2bf(float f) { return (__bf16)f; }

// ---------------------------------------------------------------------------
// Weight prep: dst[o][k] = (bf16) src[k][o]   (src is [K][64] row-major f32)
// ---------------------------------------------------------------------------
__global__ void transpose_cvt_bf16(const float* __restrict__ src,
                                   __bf16* __restrict__ dst, int K) {
  int tid = blockIdx.x * blockDim.x + threadIdx.x;
  int total = K * DIM;
  if (tid >= total) return;
  int o = tid / K;
  int k = tid - o * K;
  dst[tid] = f2bf(src[(size_t)k * DIM + o]);
}

// ---------------------------------------------------------------------------
// Edge scatter: one wave (32 lanes) per edge; 2 floats per lane.
// acc[(dst*KTOT + kidx)*64 + d] += b * x[src][d] for the 4 bilinear buckets.
// ---------------------------------------------------------------------------
template <int KK>
__global__ __launch_bounds__(256)
void spline_scatter(const float* __restrict__ x,
                    const int* __restrict__ eidx,       // [2][E]
                    const float* __restrict__ attr,     // [E][2]
                    float* __restrict__ acc,
                    float* __restrict__ deg,
                    int compute_deg) {
  const int KTOT = KK * KK;
  const int lane = threadIdx.x & 31;
  const int wave = threadIdx.x >> 5;
  const int e = blockIdx.x * 8 + wave;
  if (e >= N_EDGES) return;

  const int src = eidx[e];
  const int dst = eidx[N_EDGES + e];
  const float2 a2 = *(const float2*)(attr + 2 * (size_t)e);

  const float vx = a2.x * (float)(KK - 1);
  const float vy = a2.y * (float)(KK - 1);
  const float fx0 = floorf(vx), fy0 = floorf(vy);
  const int ix = (int)fx0, iy = (int)fy0;
  const float fx = vx - fx0, fy = vy - fy0;

  const float2 xs = *(const float2*)(x + (size_t)src * DIM + lane * 2);
  float* accbase = acc + (size_t)dst * KTOT * DIM + lane * 2;

#pragma unroll
  for (int bx = 0; bx < 2; ++bx) {
    const int kx = min(max(ix + bx, 0), KK - 1);
    const float wx = bx ? fx : 1.0f - fx;
#pragma unroll
    for (int by = 0; by < 2; ++by) {
      const int ky = min(max(iy + by, 0), KK - 1);
      const float wy = by ? fy : 1.0f - fy;
      const float b = wx * wy;
      float* p = accbase + (size_t)(kx + ky * KK) * DIM;
      unsafeAtomicAdd(p + 0, b * xs.x);
      unsafeAtomicAdd(p + 1, b * xs.y);
    }
  }
  if (compute_deg && lane == 0) unsafeAtomicAdd(deg + dst, 1.0f);
}

// ---------------------------------------------------------------------------
// Fragment helpers (contiguous k order used by this code, not the ISA table order; wave32)
//   A (16x32 bf16): lane<16 row=lane holds K = 0..15 ; lane>=16 holds
//                   K = 16..31  => K(e,half) = e + 16*half
//   B (32x16 bf16): lane N = lane&15, lanes0-15 K=0..15, lanes16-31 K=16..31
//   C/D (16x16 f32): elem j -> row j + 8*(lane>>4), col lane&15
// ---------------------------------------------------------------------------
__device__ __forceinline__ v16bf load_a_frag_lds(const float* As, int mrow,
                                                 int half) {
  v16bf a;
#pragma unroll
  for (int e = 0; e < 16; ++e) {
    const int k = e + half * 16;
    a[e] = f2bf(As[mrow * 33 + k]);
  }
  return a;
}

// ---------------------------------------------------------------------------
// Dense contraction per layer:
//   out = ELU( (acc[n] . W)/max(deg,1) + x[n] . root + bias )
// Block = 128 threads = 4 waves; 16 output rows x 64 output cols.
// ---------------------------------------------------------------------------
template <int KTOT>
__global__ __launch_bounds__(128)
void spline_gemm(const float* __restrict__ A,      // [N][KTOT*64]
                 const float* __restrict__ X,      // [N][64] root input
                 const __bf16* __restrict__ Bt,    // [64][KTOT*64] (W^T, bf16)
                 const __bf16* __restrict__ Btr,   // [64][64]      (root^T)
                 const float* __restrict__ deg,
                 const float* __restrict__ bias,
                 float* __restrict__ out) {
  const int KD = KTOT * DIM;
  __shared__ float As[16 * 33];

  const int m0 = blockIdx.x * 16;
  const int lane = threadIdx.x & 31;
  const int wave = threadIdx.x >> 5;
  const int n0 = wave * 16;
  const int half = lane >> 4;
  const int mrow = lane & 15;
  const int n = n0 + (lane & 15);

  // ---- message term: acc @ W -------------------------------------------
  v8f cmsg = {};
  for (int kb = 0; kb < KD; kb += 32) {
    __syncthreads();
    {
      const int idx = threadIdx.x * 4;  // 128 thr x 4 floats = 16x32 tile
      const int r = idx >> 5, c = idx & 31;
      const float4 v = *(const float4*)(A + (size_t)(m0 + r) * KD + kb + c);
      As[r * 33 + c + 0] = v.x;
      As[r * 33 + c + 1] = v.y;
      As[r * 33 + c + 2] = v.z;
      As[r * 33 + c + 3] = v.w;
    }
    __syncthreads();
    const v16bf afrag = load_a_frag_lds(As, mrow, half);
    const v16bf bfrag = *(const v16bf*)(Bt + (size_t)n * KD + kb + half * 16);
    cmsg = __builtin_amdgcn_wmma_f32_16x16x32_bf16(false, afrag, false, bfrag,
                                                   (short)0, cmsg, false, false);
  }

  // ---- root term: x @ root (not divided by deg) ------------------------
  v8f croot = {};
  for (int kb = 0; kb < DIM; kb += 32) {
    __syncthreads();
    {
      const int idx = threadIdx.x * 4;
      const int r = idx >> 5, c = idx & 31;
      const float4 v = *(const float4*)(X + (size_t)(m0 + r) * DIM + kb + c);
      As[r * 33 + c + 0] = v.x;
      As[r * 33 + c + 1] = v.y;
      As[r * 33 + c + 2] = v.z;
      As[r * 33 + c + 3] = v.w;
    }
    __syncthreads();
    const v16bf afrag = load_a_frag_lds(As, mrow, half);
    const v16bf bfrag = *(const v16bf*)(Btr + (size_t)n * DIM + kb + half * 16);
    croot = __builtin_amdgcn_wmma_f32_16x16x32_bf16(false, afrag, false, bfrag,
                                                    (short)0, croot, false, false);
  }

  // ---- epilogue: /deg, +bias, ELU --------------------------------------
  const float bn = bias[n];
#pragma unroll
  for (int j = 0; j < 8; ++j) {
    const int m = m0 + j + 8 * half;
    const float dg = fmaxf(deg[m], 1.0f);
    float v = cmsg[j] / dg + croot[j] + bn;
    v = (v > 0.0f) ? v : (expf(v) - 1.0f);
    out[(size_t)m * DIM + n] = v;
  }
}

// ---------------------------------------------------------------------------
// MLP head: relu(relu(h @ mlp1_W + b1) @ mlp2_W + b2) -> [N, 2]
// ---------------------------------------------------------------------------
__global__ __launch_bounds__(128)
void mlp_head(const float* __restrict__ H,      // [N][64]
              const __bf16* __restrict__ Bt1,   // [64][64] mlp1_W^T bf16
              const float* __restrict__ b1,
              const float* __restrict__ W2,     // [64][2] f32
              const float* __restrict__ b2,
              float* __restrict__ out) {        // [N][2]
  __shared__ float As[16 * 33];
  __shared__ float Ts[16 * 65];

  const int m0 = blockIdx.x * 16;
  const int lane = threadIdx.x & 31;
  const int wave = threadIdx.x >> 5;
  const int n0 = wave * 16;
  const int half = lane >> 4;
  const int mrow = lane & 15;
  const int n = n0 + (lane & 15);

  v8f c = {};
  for (int kb = 0; kb < DIM; kb += 32) {
    __syncthreads();
    {
      const int idx = threadIdx.x * 4;
      const int r = idx >> 5, cc = idx & 31;
      const float4 v = *(const float4*)(H + (size_t)(m0 + r) * DIM + kb + cc);
      As[r * 33 + cc + 0] = v.x;
      As[r * 33 + cc + 1] = v.y;
      As[r * 33 + cc + 2] = v.z;
      As[r * 33 + cc + 3] = v.w;
    }
    __syncthreads();
    const v16bf afrag = load_a_frag_lds(As, mrow, half);
    const v16bf bfrag = *(const v16bf*)(Bt1 + (size_t)n * DIM + kb + half * 16);
    c = __builtin_amdgcn_wmma_f32_16x16x32_bf16(false, afrag, false, bfrag,
                                                (short)0, c, false, false);
  }

  const float bn = b1[n];
#pragma unroll
  for (int j = 0; j < 8; ++j) {
    const int rloc = j + 8 * half;
    Ts[rloc * 65 + n] = fmaxf(c[j] + bn, 0.0f);
  }
  __syncthreads();

  if (threadIdx.x < 32) {
    const int r = threadIdx.x >> 1;
    const int col = threadIdx.x & 1;
    float s = b2[col];
#pragma unroll 8
    for (int k = 0; k < 64; ++k) s += Ts[r * 65 + k] * W2[k * 2 + col];
    out[(size_t)(m0 + r) * 2 + col] = fmaxf(s, 0.0f);
  }
}

// ---------------------------------------------------------------------------
extern "C" void kernel_launch(void* const* d_in, const int* in_sizes, int n_in,
                              void* d_out, int out_size, void* d_ws,
                              size_t ws_size, hipStream_t stream) {
  const float* x    = (const float*)d_in[0];
  const int*   eidx = (const int*)d_in[1];   // edge_index [2][E]
  const float* attr = (const float*)d_in[2]; // [E][2]
  const float* W1   = (const float*)d_in[3]; // [9][64][64]
  const float* r1   = (const float*)d_in[4];
  const float* bb1  = (const float*)d_in[5];
  const float* W2   = (const float*)d_in[6]; // [25][64][64]
  const float* r2   = (const float*)d_in[7];
  const float* bb2  = (const float*)d_in[8];
  const float* m1W  = (const float*)d_in[9];
  const float* m1b  = (const float*)d_in[10];
  const float* m2W  = (const float*)d_in[11];
  const float* m2b  = (const float*)d_in[12];
  float* out = (float*)d_out;

  char* ws = (char*)d_ws;
  size_t off = 0;
  auto carve = [&](size_t bytes) -> void* {
    void* p = ws + off;
    off += (bytes + 255) & ~(size_t)255;
    return p;
  };
  float*  acc  = (float*)carve((size_t)N_NODES * 25 * DIM * 4); // 320 MB
  float*  deg  = (float*)carve((size_t)N_NODES * 4);
  float*  h1   = (float*)carve((size_t)N_NODES * DIM * 4);
  float*  h2   = (float*)carve((size_t)N_NODES * DIM * 4);
  __bf16* Bt1  = (__bf16*)carve((size_t)DIM * 576 * 2);
  __bf16* Bt1r = (__bf16*)carve((size_t)DIM * DIM * 2);
  __bf16* Bt2  = (__bf16*)carve((size_t)DIM * 1600 * 2);
  __bf16* Bt2r = (__bf16*)carve((size_t)DIM * DIM * 2);
  __bf16* Btm1 = (__bf16*)carve((size_t)DIM * DIM * 2);

  // --- weight prep (bf16, transposed so B fragments are contiguous) ------
  transpose_cvt_bf16<<<(576 * 64 + 255) / 256, 256, 0, stream>>>(W1, Bt1, 576);
  transpose_cvt_bf16<<<(64 * 64 + 255) / 256, 256, 0, stream>>>(r1, Bt1r, 64);
  transpose_cvt_bf16<<<(1600 * 64 + 255) / 256, 256, 0, stream>>>(W2, Bt2, 1600);
  transpose_cvt_bf16<<<(64 * 64 + 255) / 256, 256, 0, stream>>>(r2, Bt2r, 64);
  transpose_cvt_bf16<<<(64 * 64 + 255) / 256, 256, 0, stream>>>(m1W, Btm1, 64);

  // --- layer 1 (K=3, Ktot=9) --------------------------------------------
  hipMemsetAsync(acc, 0, (size_t)N_NODES * 9 * DIM * 4, stream);
  hipMemsetAsync(deg, 0, (size_t)N_NODES * 4, stream);
  spline_scatter<3><<<N_EDGES / 8, 256, 0, stream>>>(x, eidx, attr, acc, deg, 1);
  spline_gemm<9><<<N_NODES / 16, 128, 0, stream>>>(acc, x, Bt1, Bt1r, deg, bb1, h1);

  // --- layer 2 (K=5, Ktot=25); deg identical (same edge list) -----------
  hipMemsetAsync(acc, 0, (size_t)N_NODES * 25 * DIM * 4, stream);
  spline_scatter<5><<<N_EDGES / 8, 256, 0, stream>>>(h1, eidx, attr, acc, deg, 0);
  spline_gemm<25><<<N_NODES / 16, 128, 0, stream>>>(acc, h1, Bt2, Bt2r, deg, bb2, h2);

  // --- MLP head ----------------------------------------------------------
  mlp_head<<<N_NODES / 16, 128, 0, stream>>>(h2, Btm1, m1b, m2W, m2b, out);
}
